// TopKSAE_2568390443168
// MI455X (gfx1250) — matrix operand order repaired and checked by reading, not compiled, not run on hardware
//
#include <hip/hip_runtime.h>
#include <hip/hip_bf16.h>

// ---------------------------------------------------------------------------
// TopK SAE forward for MI455X (gfx1250, wave32, WMMA + TDM).
//
//   K0  tiled transpose+convert W_enc f32[768,24576] -> bf16 W_encT[24576,768]
//       (both GEMM operands become K-contiguous -> b128 LDS fragment loads)
//   K1  per-row preprocess: mean, std(ddof=1), xn, bf16(xn - b_dec)
//   K2  encoder GEMM via v_wmma_f32_16x16x32_bf16, tiles staged by the
//       Tensor Data Mover (tensor_load_to_lds, double buffered,
//       s_wait_tensorcnt), relu, dense store into acts_topk region of d_out
//   K3  deterministic radix-select top-32 per row (+ l1/l0 partials)
//   K4  deterministic radix-select top-256 among dead features per row
//   K5  zero acts region + scatter top-32 (final sparse acts_topk output)
//   K6  sparse decode (32 nz) + b_dec, sae_out = x_rec*std+mean, l2 partials
//   K7  aux sparse decode (256 nz), aux-loss partials vs residual
//   K8  cosine penalty: 32x32 gram of gathered W_dec rows via WMMA (bf16)
//   K9  final reduction -> 7 scalars
// ---------------------------------------------------------------------------

#define ACT   768
#define DICT  24576
#define BATCH 2048
#define TOPK  32
#define TOPKAUX 256
#define NDEAD 5
#define EPSF  1e-5f

typedef __attribute__((ext_vector_type(16))) __bf16 v16bf;
typedef __attribute__((ext_vector_type(8)))  float  v8f;
typedef __attribute__((ext_vector_type(4)))  unsigned int u32x4;
typedef __attribute__((ext_vector_type(8)))  int i32x8;
typedef __attribute__((ext_vector_type(4)))  int i32x4;

#if defined(__gfx1250__) && __has_builtin(__builtin_amdgcn_tensor_load_to_lds)
#define HAVE_TDM 1
#else
#define HAVE_TDM 0
#endif

// Low 32 bits of a flat pointer into LDS == LDS byte offset (ISA 10.2).
__device__ __forceinline__ unsigned lds_off(const void* p) {
  return (unsigned)(size_t)p;
}

#if HAVE_TDM
// Issue a 2-D TDM tile load (global -> LDS). Tile is fully in-bounds, so
// tensor dims are set to the tile dims (coords are tile-start relative:
// D#.global_addr is the tile start). pad_interval/pad_amount are the raw
// encoded fields: interval 2^(v+1) DWORDs, amount (v+1) DWORDs.
__device__ __forceinline__ void tdm_load_2d(unsigned lds_byte_off,
                                            unsigned long long gaddr,
                                            unsigned tile_dim0,   // elems, contiguous
                                            unsigned tile_dim1,   // rows
                                            unsigned long long row_stride, // elems
                                            unsigned pad_interval_enc,
                                            unsigned pad_amount_enc) {
  u32x4 g0;
  g0[0] = 1u;                                        // count=1, user descriptor
  g0[1] = lds_byte_off;                              // lds_addr
  g0[2] = (unsigned)(gaddr & 0xFFFFFFFFull);         // global_addr[31:0]
  g0[3] = (unsigned)((gaddr >> 32) & 0x01FFFFFFull)  // global_addr[56:32]
        | (2u << 30);                                // type=2 ("image")
  i32x8 g1;
  g1[0] = (int)((1u << 16)                           // data_size=1 -> 2 bytes
              | (1u << 20)                           // pad_enable
              | (pad_interval_enc << 22)
              | (pad_amount_enc << 25));
  g1[1] = (int)((tile_dim0 & 0xFFFFu) << 16);        // tensor_dim0[15:0]
  g1[2] = (int)(((tile_dim0 >> 16) & 0xFFFFu)        // tensor_dim0[31:16]
              | ((tile_dim1 & 0xFFFFu) << 16));      // tensor_dim1[15:0]
  g1[3] = (int)(((tile_dim1 >> 16) & 0xFFFFu)        // tensor_dim1[31:16]
              | ((tile_dim0 & 0xFFFFu) << 16));      // tile_dim0
  g1[4] = (int)(tile_dim1 & 0xFFFFu);                // tile_dim1 (tile_dim2=0)
  g1[5] = (int)(row_stride & 0xFFFFFFFFull);         // tensor_dim0_stride[31:0]
  g1[6] = (int)((row_stride >> 32) & 0xFFFFull);     // stride[47:32], dim1_stride=0
  g1[7] = 0;
  i32x4 gz = {0, 0, 0, 0};
#if __clang_major__ >= 23
  i32x8 gz8 = {0, 0, 0, 0, 0, 0, 0, 0};
  __builtin_amdgcn_tensor_load_to_lds(g0, g1, gz, gz, gz8, 0);
#else
  __builtin_amdgcn_tensor_load_to_lds(g0, g1, gz, gz, 0);
#endif
}
#endif // HAVE_TDM

// ---------------- K0: transpose + convert W_enc -> bf16 W_encT -------------
// W_enc is [768][24576] f32 row-major; produce WT [24576][768] bf16 so the
// GEMM B operand is K-contiguous (b128 LDS fragment loads, TDM-friendly).
__global__ __launch_bounds__(256) void k_transpose_bf16(const float* __restrict__ W,
                                                        __bf16* __restrict__ WT) {
  __shared__ float tile[32][33];
  const int tx = threadIdx.x & 31, ty = threadIdx.x >> 5;   // 32x8
  const int n0 = blockIdx.x * 32, k0 = blockIdx.y * 32;
  #pragma unroll
  for (int s = 0; s < 4; ++s) {
    int r = ty + 8 * s;
    tile[r][tx] = W[(size_t)(k0 + r) * DICT + n0 + tx];
  }
  __syncthreads();
  #pragma unroll
  for (int s = 0; s < 4; ++s) {
    int r = ty + 8 * s;                       // n within tile
    WT[(size_t)(n0 + r) * ACT + k0 + tx] = (__bf16)tile[tx][r];
  }
}

// ------------------------------- K1: preprocess ----------------------------
__global__ __launch_bounds__(256) void k_preprocess(const float* __restrict__ x,
                                                    const float* __restrict__ b_dec,
                                                    float*  __restrict__ xn,
                                                    __bf16* __restrict__ abf,
                                                    float*  __restrict__ meanv,
                                                    float*  __restrict__ stdv) {
  __shared__ float red[256];
  const int row = blockIdx.x, tid = threadIdx.x;
  const float* xr = x + (size_t)row * ACT;

  float s = 0.f;
  for (int c = tid; c < ACT; c += 256) s += xr[c];
  red[tid] = s; __syncthreads();
  for (int o = 128; o > 0; o >>= 1) { if (tid < o) red[tid] += red[tid + o]; __syncthreads(); }
  const float mean = red[0] / (float)ACT;
  __syncthreads();

  float v = 0.f;
  for (int c = tid; c < ACT; c += 256) { float d = xr[c] - mean; v += d * d; }
  red[tid] = v; __syncthreads();
  for (int o = 128; o > 0; o >>= 1) { if (tid < o) red[tid] += red[tid + o]; __syncthreads(); }
  const float sd  = sqrtf(red[0] / (float)(ACT - 1));   // ddof=1
  const float inv = 1.f / (sd + EPSF);

  for (int c = tid; c < ACT; c += 256) {
    float xnv = (xr[c] - mean) * inv;
    xn[(size_t)row * ACT + c]  = xnv;
    abf[(size_t)row * ACT + c] = (__bf16)(xnv - b_dec[c]);
  }
  if (tid == 0) { meanv[row] = mean; stdv[row] = sd; }
}

// ------------------------------- K2: encoder GEMM --------------------------
// Block tile 64x128, 8 waves: wave(wm 0..3, wn 0..1) -> 16 rows x 4x16 cols.
// Both LDS operands are [rows][K + 8] bf16: 80-byte rows => 16B-aligned
// 8-element chunks per lane (ds_load_b128), 20-bank lane stride.
// All fragment loads are hoisted ahead of the 4 WMMAs so the ds_load_b128s
// pipeline and the matrix ops issue back-to-back.
#define BM 64
#define BN 128
#define BK 32
#define KP 8
__global__ __launch_bounds__(256) void k_encode_gemm(const __bf16* __restrict__ A,
                                                     const __bf16* __restrict__ WT,
                                                     float* __restrict__ acts) {
  __shared__ __bf16 As[2][BM][BK + KP];
  __shared__ __bf16 Bs[2][BN][BK + KP];
  const int tid = threadIdx.x;
  const int lane = tid & 31, w = tid >> 5;
  const int wm = w >> 1, wn = w & 1;
  const int m0 = blockIdx.y * BM, n0 = blockIdx.x * BN;
  const int NK = ACT / BK;                       // 24 k-steps

  v8f acc[4] = {};

#if HAVE_TDM
  const unsigned long long aA = (unsigned long long)(size_t)(A  + (size_t)m0 * ACT);
  const unsigned long long aB = (unsigned long long)(size_t)(WT + (size_t)n0 * ACT);
  const unsigned ldsA[2] = { lds_off(&As[0][0][0]), lds_off(&As[1][0][0]) };
  const unsigned ldsB[2] = { lds_off(&Bs[0][0][0]), lds_off(&Bs[1][0][0]) };
  // pad: after 16 DWORDs (enc 3) insert 4 DWORDs (enc 3) -> 40-bf16 rows.
  if (w == 0) {
    tdm_load_2d(ldsA[0], aA, BK, BM, ACT, 3, 3);
    tdm_load_2d(ldsB[0], aB, BK, BN, ACT, 3, 3);
    __builtin_amdgcn_s_wait_tensorcnt(0);
  }
  __syncthreads();
#endif

  for (int i = 0; i < NK; ++i) {
#if HAVE_TDM
    const int cur = i & 1;
    if (w == 0 && i + 1 < NK) {               // prefetch next tiles (overlaps WMMA)
      const unsigned long long koff = (unsigned long long)(i + 1) * BK * sizeof(__bf16);
      tdm_load_2d(ldsA[cur ^ 1], aA + koff, BK, BM, ACT, 3, 3);
      tdm_load_2d(ldsB[cur ^ 1], aB + koff, BK, BN, ACT, 3, 3);
    }
#else
    const int cur = 0;
    const int k0 = i * BK;
    #pragma unroll
    for (int t = 0; t < (BM * BK) / 256; ++t) {
      int ii = tid + t * 256, r = ii >> 5, c = ii & 31;
      As[0][r][c] = A[(size_t)(m0 + r) * ACT + k0 + c];
    }
    #pragma unroll
    for (int t = 0; t < (BN * BK) / 256; ++t) {
      int ii = tid + t * 256, n = ii >> 5, c = ii & 31;
      Bs[0][n][c] = WT[(size_t)(n0 + n) * ACT + k0 + c];
    }
    __syncthreads();
#endif

    // Hoist ALL fragment loads (A + 4xB) so ds_load_b128s pipeline together.
    // A fragment: lane m=l&15; K split over lane halves,
    // 16 contiguous K elems (32B) per lane.
    v16bf af;
    {
      const __bf16* arow = &As[cur][wm * 16 + (lane & 15)][(lane < 16) ? 0 : 16];
      #pragma unroll
      for (int j = 0; j < 16; ++j) af[j] = arow[j];
    }
    // B fragments: lane n=l&15; 16 contiguous K elems (32B) per lane.
    v16bf bfr[4];
    #pragma unroll
    for (int t = 0; t < 4; ++t) {
      const __bf16* brow = &Bs[cur][wn * 64 + t * 16 + (lane & 15)][(lane < 16) ? 0 : 16];
      #pragma unroll
      for (int j = 0; j < 16; ++j) bfr[t][j] = brow[j];
    }
    #pragma unroll
    for (int t = 0; t < 4; ++t) {
      acc[t] = __builtin_amdgcn_wmma_f32_16x16x32_bf16(false, af, false, bfr[t],
                                                       (short)0, acc[t], false, false);
    }

#if HAVE_TDM
    if (w == 0 && i + 1 < NK) __builtin_amdgcn_s_wait_tensorcnt(0);
#endif
    __syncthreads();
  }

  // C layout: M = r + 8*laneHi, N = lane&15
  const int rbase = m0 + wm * 16 + ((lane < 16) ? 0 : 8);
  #pragma unroll
  for (int t = 0; t < 4; ++t) {
    const int gn = n0 + wn * 64 + t * 16 + (lane & 15);
    #pragma unroll
    for (int r = 0; r < 8; ++r) {
      float vv = acc[t][r];
      acts[(size_t)(rbase + r) * DICT + gn] = vv > 0.f ? vv : 0.f;   // relu
    }
  }
}

// --------------------- K3/K4: deterministic radix top-K --------------------
template <int K, bool MASKED, bool STATS>
__global__ __launch_bounds__(256) void k_radix_topk(const float* __restrict__ acts,
                                                    const int* __restrict__ nba,
                                                    int*   __restrict__ idx_all,
                                                    float* __restrict__ val_all,
                                                    float* __restrict__ l1p,
                                                    float* __restrict__ l0p) {
  __shared__ unsigned hist[256];
  __shared__ unsigned offG[257], offE[257];
  __shared__ unsigned bc_prefix, bc_want;
  __shared__ float red[256];

  const int row = blockIdx.x, tid = threadIdx.x;
  const float* ar = acts + (size_t)row * DICT;
  int*   idx_out = idx_all + (size_t)row * K;
  float* val_out = val_all + (size_t)row * K;

  unsigned prefix = 0;
  int want = K;
  for (int shift = 24; shift >= 0; shift -= 8) {
    hist[tid] = 0; __syncthreads();
    const unsigned himask = (shift == 24) ? 0u : (0xFFFFFFFFu << (shift + 8));
    for (int i = tid; i < DICT; i += 256) {
      if constexpr (MASKED) { if (nba[i] < NDEAD) continue; }
      unsigned u = __float_as_uint(ar[i]);
      if ((u & himask) == (prefix & himask))
        atomicAdd(&hist[(u >> shift) & 255], 1u);
    }
    __syncthreads();
    if (tid == 0) {
      unsigned cum = 0; int chosen = 0, rem = want;
      for (int d = 255; d >= 0; --d) {
        unsigned c = hist[d];
        if (cum + c >= (unsigned)want || d == 0) { chosen = d; rem = want - (int)cum; break; }
        cum += c;
      }
      bc_prefix = prefix | ((unsigned)chosen << shift);
      bc_want   = (unsigned)(rem > 0 ? rem : 0);
    }
    __syncthreads();
    prefix = bc_prefix; want = (int)bc_want;
    __syncthreads();
  }

  const unsigned T = prefix;
  for (int j = tid; j < K; j += 256) { idx_out[j] = 0; val_out[j] = 0.f; }

  const int CH = DICT / 256;
  const int base = tid * CH;
  unsigned cG = 0, cE = 0;
  for (int i = base; i < base + CH; ++i) {
    if constexpr (MASKED) { if (nba[i] < NDEAD) continue; }
    unsigned u = __float_as_uint(ar[i]);
    if (u > T) cG++; else if (u == T) cE++;
  }
  offG[tid] = cG; offE[tid] = cE;
  __syncthreads();
  if (tid == 0) {
    unsigned aG = 0, aE = 0;
    for (int t = 0; t < 256; ++t) {
      unsigned g = offG[t]; offG[t] = aG; aG += g;
      unsigned e = offE[t]; offE[t] = aE; aE += e;
    }
    offG[256] = aG; offE[256] = aE;
  }
  __syncthreads();
  const unsigned Gtotal = offG[256];
  unsigned pG = offG[tid];
  unsigned pE = Gtotal + offE[tid];
  for (int i = base; i < base + CH; ++i) {
    if constexpr (MASKED) { if (nba[i] < NDEAD) continue; }
    float v = ar[i];
    unsigned u = __float_as_uint(v);
    if (u > T)       { if (pG < (unsigned)K) { idx_out[pG] = i; val_out[pG] = v; } pG++; }
    else if (u == T) { if (pE < (unsigned)K) { idx_out[pE] = i; val_out[pE] = v; } pE++; }
  }
  __syncthreads();

  if constexpr (STATS) {
    float s1 = 0.f, s0 = 0.f;
    for (int j = tid; j < K; j += 256) { float v = val_out[j]; s1 += v; if (v > 0.f) s0 += 1.f; }
    red[tid] = s1; __syncthreads();
    for (int o = 128; o > 0; o >>= 1) { if (tid < o) red[tid] += red[tid + o]; __syncthreads(); }
    float t1 = red[0]; __syncthreads();
    red[tid] = s0; __syncthreads();
    for (int o = 128; o > 0; o >>= 1) { if (tid < o) red[tid] += red[tid + o]; __syncthreads(); }
    if (tid == 0) { l1p[row] = t1; l0p[row] = red[0]; }
  }
}

// ------------------------------- K5: sparsify ------------------------------
__global__ __launch_bounds__(256) void k_zero(float* __restrict__ p, size_t n) {
  size_t i = (size_t)blockIdx.x * blockDim.x + threadIdx.x;
  size_t stride = (size_t)gridDim.x * blockDim.x;
  for (; i < n; i += stride) p[i] = 0.f;
}
__global__ __launch_bounds__(256) void k_scatter(float* __restrict__ acts,
                                                 const int* __restrict__ idx,
                                                 const float* __restrict__ val) {
  int i = blockIdx.x * blockDim.x + threadIdx.x;
  if (i < BATCH * TOPK) {
    int row = i / TOPK;
    acts[(size_t)row * DICT + idx[i]] = val[i];
  }
}

// ------------------------------- K6: decode + l2 ---------------------------
__global__ __launch_bounds__(256) void k_decode(const float* __restrict__ Wd,
                                                const float* __restrict__ bdec,
                                                const int* __restrict__ idx,
                                                const float* __restrict__ val,
                                                const float* __restrict__ xn,
                                                const float* __restrict__ meanv,
                                                const float* __restrict__ stdv,
                                                float* __restrict__ xrec,
                                                float* __restrict__ sae_out,
                                                float* __restrict__ l2p) {
  __shared__ float red[256];
  __shared__ int   sidx[TOPK];
  __shared__ float sval[TOPK];
  const int row = blockIdx.x, tid = threadIdx.x;
  if (tid < TOPK) { sidx[tid] = idx[row * TOPK + tid]; sval[tid] = val[row * TOPK + tid]; }
  __syncthreads();

  float acc[3];
  #pragma unroll
  for (int q = 0; q < 3; ++q) acc[q] = bdec[tid + q * 256];
  for (int j = 0; j < TOPK; ++j) {
    const float* wr = Wd + (size_t)sidx[j] * ACT;
    const float  v  = sval[j];
    #pragma unroll
    for (int q = 0; q < 3; ++q) acc[q] += v * wr[tid + q * 256];
  }
  const float mean = meanv[row], sd = stdv[row];
  float l2 = 0.f;
  #pragma unroll
  for (int q = 0; q < 3; ++q) {
    int c = tid + q * 256;
    xrec[(size_t)row * ACT + c]    = acc[q];
    sae_out[(size_t)row * ACT + c] = acc[q] * sd + mean;   // postprocess
    float d = acc[q] - xn[(size_t)row * ACT + c];
    l2 += d * d;
  }
  red[tid] = l2; __syncthreads();
  for (int o = 128; o > 0; o >>= 1) { if (tid < o) red[tid] += red[tid + o]; __syncthreads(); }
  if (tid == 0) l2p[row] = red[0];
}

// ------------------------------- K7: aux loss ------------------------------
__global__ __launch_bounds__(256) void k_auxdecode(const float* __restrict__ Wd,
                                                   const int* __restrict__ idx,
                                                   const float* __restrict__ val,
                                                   const float* __restrict__ xn,
                                                   const float* __restrict__ xrec,
                                                   float* __restrict__ auxp) {
  __shared__ float red[256];
  __shared__ int   sidx[TOPKAUX];
  __shared__ float sval[TOPKAUX];
  const int row = blockIdx.x, tid = threadIdx.x;
  sidx[tid] = idx[(size_t)row * TOPKAUX + tid];
  sval[tid] = val[(size_t)row * TOPKAUX + tid];
  __syncthreads();

  float acc[3] = {0.f, 0.f, 0.f};
  for (int j = 0; j < TOPKAUX; ++j) {
    const float* wr = Wd + (size_t)sidx[j] * ACT;
    const float  v  = sval[j];
    #pragma unroll
    for (int q = 0; q < 3; ++q) acc[q] += v * wr[tid + q * 256];
  }
  float s = 0.f;
  #pragma unroll
  for (int q = 0; q < 3; ++q) {
    int c = tid + q * 256;
    float resid = xn[(size_t)row * ACT + c] - xrec[(size_t)row * ACT + c];
    float d = acc[q] - resid;
    s += d * d;
  }
  red[tid] = s; __syncthreads();
  for (int o = 128; o > 0; o >>= 1) { if (tid < o) red[tid] += red[tid + o]; __syncthreads(); }
  if (tid == 0) auxp[row] = red[0];
}

// ------------------------------- K8: cosine penalty ------------------------
__global__ __launch_bounds__(256) void k_cosine(const float* __restrict__ Wd,
                                                const int* __restrict__ idx,
                                                float* __restrict__ cosp) {
  __shared__ __bf16 D[32][ACT + 8];
  __shared__ float red[256];
  const int row = blockIdx.x, tid = threadIdx.x;
  const int lane = tid & 31, w = tid >> 5;

  for (int j = 0; j < 32; ++j) {
    const float* wr = Wd + (size_t)idx[row * TOPK + j] * ACT;
    for (int c = tid; c < ACT; c += 256) D[j][c] = (__bf16)wr[c];
  }
  __syncthreads();

  float local = 0.f;
  if (w < 4) {
    const int ti = w >> 1, tj = w & 1;
    v8f acc = {};
    for (int kk = 0; kk < ACT; kk += 32) {
      v16bf af, bfr;
      const int m  = ti * 16 + (lane & 15);
      const int ka = kk + ((lane < 16) ? 0 : 16);
      #pragma unroll
      for (int j = 0; j < 16; ++j) af[j] = D[m][ka + j];
      const int n  = tj * 16 + (lane & 15);
      const int kb = kk + ((lane < 16) ? 0 : 16);
      #pragma unroll
      for (int j = 0; j < 16; ++j) bfr[j] = D[n][kb + j];
      acc = __builtin_amdgcn_wmma_f32_16x16x32_bf16(false, af, false, bfr,
                                                    (short)0, acc, false, false);
    }
    #pragma unroll
    for (int r = 0; r < 8; ++r) {
      const int i    = ti * 16 + r + ((lane < 16) ? 0 : 8);
      const int jcol = tj * 16 + (lane & 15);
      if (i != jcol) local += fabsf(acc[r]);
    }
  }
  red[tid] = local; __syncthreads();
  for (int o = 128; o > 0; o >>= 1) { if (tid < o) red[tid] += red[tid + o]; __syncthreads(); }
  if (tid == 0) cosp[row] = red[0];
}

// ------------------------------- K9: final scalars -------------------------
__global__ __launch_bounds__(256) void k_final(const float* __restrict__ l2p,
                                               const float* __restrict__ l1p,
                                               const float* __restrict__ l0p,
                                               const float* __restrict__ auxp,
                                               const float* __restrict__ cosp,
                                               float* __restrict__ out) {
  __shared__ float red[256];
  const int tid = threadIdx.x;
  float s[5] = {0.f, 0.f, 0.f, 0.f, 0.f};
  for (int i = tid; i < BATCH; i += 256) {
    s[0] += l2p[i]; s[1] += l1p[i]; s[2] += l0p[i]; s[3] += auxp[i]; s[4] += cosp[i];
  }
  float tot[5];
  for (int q = 0; q < 5; ++q) {
    red[tid] = s[q]; __syncthreads();
    for (int o = 128; o > 0; o >>= 1) { if (tid < o) red[tid] += red[tid + o]; __syncthreads(); }
    tot[q] = red[0]; __syncthreads();
  }
  if (tid == 0) {
    float l2  = tot[0] / ((float)BATCH * ACT);
    float l1n = tot[1] / (float)BATCH;
    float l1l = 1.0e-4f * l1n;
    float l0  = tot[2] / (float)BATCH;
    float aux = 0.03125f * (tot[3] / ((float)BATCH * ACT));
    float ms  = tot[4] / ((float)BATCH * TOPK * (TOPK - 1));
    out[0] = l2 + l1l + aux + 0.01f * ms;   // loss
    out[1] = l2; out[2] = l1l; out[3] = l0; out[4] = l1n; out[5] = aux; out[6] = ms;
  }
}

// ------------------------------- launch ------------------------------------
extern "C" void kernel_launch(void* const* d_in, const int* in_sizes, int n_in,
                              void* d_out, int out_size, void* d_ws, size_t ws_size,
                              hipStream_t stream) {
  (void)in_sizes; (void)n_in; (void)out_size; (void)ws_size;
  const float* x     = (const float*)d_in[0];
  const float* W_enc = (const float*)d_in[1];
  const float* W_dec = (const float*)d_in[2];
  const float* b_dec = (const float*)d_in[3];
  const int*   nba   = (const int*)d_in[4];

  char* ws = (char*)d_ws;
  size_t off = 0;
  auto alloc = [&](size_t bytes) -> void* {
    void* p = ws + off;
    off = (off + bytes + 255) & ~(size_t)255;
    return p;
  };
  // ~58.3 MB total workspace
  __bf16* WTbf   = (__bf16*)alloc((size_t)ACT * DICT * sizeof(__bf16));   // W_enc^T
  float*  xn     = (float*)alloc((size_t)BATCH * ACT * sizeof(float));
  __bf16* Abf    = (__bf16*)alloc((size_t)BATCH * ACT * sizeof(__bf16));
  float*  meanv  = (float*)alloc(BATCH * sizeof(float));
  float*  stdv   = (float*)alloc(BATCH * sizeof(float));
  int*    idx32  = (int*)alloc((size_t)BATCH * TOPK * sizeof(int));
  float*  val32  = (float*)alloc((size_t)BATCH * TOPK * sizeof(float));
  int*    idx256 = (int*)alloc((size_t)BATCH * TOPKAUX * sizeof(int));
  float*  val256 = (float*)alloc((size_t)BATCH * TOPKAUX * sizeof(float));
  float*  xrec   = (float*)alloc((size_t)BATCH * ACT * sizeof(float));
  float*  l2p    = (float*)alloc(BATCH * sizeof(float));
  float*  l1p    = (float*)alloc(BATCH * sizeof(float));
  float*  l0p    = (float*)alloc(BATCH * sizeof(float));
  float*  auxp   = (float*)alloc(BATCH * sizeof(float));
  float*  cosp   = (float*)alloc(BATCH * sizeof(float));

  float* sae_out = (float*)d_out;
  float* acts    = sae_out + (size_t)BATCH * ACT;         // dense, then sparsified
  float* scal    = acts + (size_t)BATCH * DICT;           // 7 scalars

  k_transpose_bf16<<<dim3(DICT / 32, ACT / 32), 256, 0, stream>>>(W_enc, WTbf);
  k_preprocess<<<BATCH, 256, 0, stream>>>(x, b_dec, xn, Abf, meanv, stdv);
  k_encode_gemm<<<dim3(DICT / BN, BATCH / BM), 256, 0, stream>>>(Abf, WTbf, acts);
  k_radix_topk<TOPK, false, true><<<BATCH, 256, 0, stream>>>(acts, nullptr, idx32, val32, l1p, l0p);
  k_radix_topk<TOPKAUX, true, false><<<BATCH, 256, 0, stream>>>(acts, nba, idx256, val256, nullptr, nullptr);
  k_zero<<<8192, 256, 0, stream>>>(acts, (size_t)BATCH * DICT);
  k_scatter<<<(BATCH * TOPK + 255) / 256, 256, 0, stream>>>(acts, idx32, val32);
  k_decode<<<BATCH, 256, 0, stream>>>(W_dec, b_dec, idx32, val32, xn, meanv, stdv, xrec, sae_out, l2p);
  k_auxdecode<<<BATCH, 256, 0, stream>>>(W_dec, idx256, val256, xn, xrec, auxp);
  k_cosine<<<BATCH, 256, 0, stream>>>(W_dec, idx32, cosp);
  k_final<<<1, 256, 0, stream>>>(l2p, l1p, l0p, auxp, cosp, scal);
}
